// GATRegressor_21758304321636
// MI455X (gfx1250) — compile-verified, matrix operand order checked by reading, not run on hardware
//
#include <hip/hip_runtime.h>
#include <hip/hip_bf16.h>

// ---------------------------------------------------------------------------
// GATRegressor on MI455X (gfx1250).
//   GEMMs  : bf16 WMMA 16x16x32, one 16x16 tile per wave32.
//   Edges  : gather / segment-softmax / scatter with f32 atomics (L2-resident:
//            node-feature buffers are ~51MB each << 192MB L2).
//   Norm   : fused column reductions + elementwise (emits bf16 for next GEMM).
// ---------------------------------------------------------------------------

typedef __attribute__((ext_vector_type(16))) __bf16 v16bf;
typedef __attribute__((ext_vector_type(8)))  __bf16 bf16x8;
typedef __attribute__((ext_vector_type(8)))  float  v8f;

#define HC 256   // H*C
#define HEADS 4
#define CH 64

// order-preserving uint encoding of float (for atomic segment-max)
#define ORD_NEG_INF 0x007FFFFFu   // ordEnc(-inf)

__device__ __forceinline__ void atomicAddF32(float* p, float v) {
    __hip_atomic_fetch_add(p, v, __ATOMIC_RELAXED, __HIP_MEMORY_SCOPE_AGENT);
}
__device__ __forceinline__ unsigned ordEnc(float x) {
    unsigned u = __float_as_uint(x);
    return (u & 0x80000000u) ? ~u : (u | 0x80000000u);
}
__device__ __forceinline__ float ordDec(unsigned u) {
    return (u & 0x80000000u) ? __uint_as_float(u & 0x7FFFFFFFu) : __uint_as_float(~u);
}

// ---------------------------------------------------------------- fill utils
__global__ void fill_f32(float* p, float v, int n) {
    int i = blockIdx.x * blockDim.x + threadIdx.x;
    if (i < n) p[i] = v;
}
__global__ void fill_u32(unsigned* p, unsigned v, int n) {
    int i = blockIdx.x * blockDim.x + threadIdx.x;
    if (i < n) p[i] = v;
}

// ------------------------------------------------------------- conversions
__global__ void cvt_f32_bf16(const float* __restrict__ in, __bf16* __restrict__ out, int n) {
    int i = blockIdx.x * blockDim.x + threadIdx.x;
    if (i < n) out[i] = (__bf16)in[i];
}
// W [K,Nc] row-major f32  ->  Wt [Nc,K] row-major bf16
__global__ void wt_bf16(const float* __restrict__ W, __bf16* __restrict__ Wt, int K, int Nc) {
    int i = blockIdx.x * blockDim.x + threadIdx.x;
    if (i >= K * Nc) return;
    int k = i / Nc, n = i % Nc;
    Wt[(size_t)n * K + k] = (__bf16)W[i];
}

// ------------------------------------------------------------ WMMA GEMM
// C[M,256] = A[M,256](bf16) * W[256,256], W given transposed (Bt[n,k] bf16).
// One 16x16 tile per wave; 4 waves/block on adjacent N tiles; K unrolled x8.
__global__ void gemm_bf16_wmma(const __bf16* __restrict__ A,
                               const __bf16* __restrict__ Bt,
                               float* __restrict__ C, int M) {
    const int lane  = threadIdx.x & 31;
    const int wave  = threadIdx.x >> 5;
    const int m0    = blockIdx.x * 16;
    const int n0    = (blockIdx.y * 4 + wave) * 16;
    const int K     = 256;
    const int rlo   = lane & 15;          // A-row / B-col inside tile
    const int khalf = (lane >> 4) * 8;    // K sub-group per 16-bit A/B striping

    const __bf16* Arow = A  + (size_t)(m0 + rlo) * K;
    const __bf16* Brow = Bt + (size_t)(n0 + rlo) * K;

    v8f acc = {};
#pragma unroll
    for (int k0 = 0; k0 < 256; k0 += 32) {
        __builtin_prefetch(Arow + k0 + 64, 0, 0);
        bf16x8 alo = *(const bf16x8*)(Arow + k0 + khalf);
        bf16x8 ahi = *(const bf16x8*)(Arow + k0 + khalf + 16);
        bf16x8 blo = *(const bf16x8*)(Brow + k0 + khalf);
        bf16x8 bhi = *(const bf16x8*)(Brow + k0 + khalf + 16);
        v16bf a, b;
#pragma unroll
        for (int i = 0; i < 8; ++i) {
            a[i] = alo[i]; a[i + 8] = ahi[i];
            b[i] = blo[i]; b[i + 8] = bhi[i];
        }
        acc = __builtin_amdgcn_wmma_f32_16x16x32_bf16(
            false, a, false, b, (short)0, acc, false, false);
    }
    const int rbase = m0 + ((lane >> 4) * 8);
    const int col   = n0 + (lane & 15);
#pragma unroll
    for (int r = 0; r < 8; ++r)
        C[(size_t)(rbase + r) * 256 + col] = acc[r];
}

// --------------------------------------------- per-node attention logits s,d
__global__ void node_attn_sd(const float* __restrict__ h,
                             const float* __restrict__ as_, const float* __restrict__ ad_,
                             float* __restrict__ s, float* __restrict__ d) {
    __shared__ float rs[HC], rd[HC];
    const int n = blockIdx.x, t = threadIdx.x;
    float hv = h[(size_t)n * HC + t];
    rs[t] = hv * as_[t];
    rd[t] = hv * ad_[t];
    __syncthreads();
    for (int off = 32; off >= 1; off >>= 1) {
        if ((t & 63) < off) { rs[t] += rs[t + off]; rd[t] += rd[t + off]; }
        __syncthreads();
    }
    if ((t & 63) == 0) {
        s[n * HEADS + (t >> 6)] = rs[t];
        d[n * HEADS + (t >> 6)] = rd[t];
    }
}

// ------------------------------------------- edge pass 1: logits + seg-max
__global__ void edge_logits_max(const int* __restrict__ srcA, const int* __restrict__ dstA,
                                const float* __restrict__ s, const float* __restrict__ d,
                                float* __restrict__ e, unsigned* __restrict__ mEnc,
                                int E, int N) {
    int i = blockIdx.x * blockDim.x + threadIdx.x;
    if (i >= E + N) return;
    int si = (i < E) ? srcA[i] : (i - E);
    int di = (i < E) ? dstA[i] : (i - E);
#pragma unroll
    for (int hh = 0; hh < HEADS; ++hh) {
        float v = s[si * HEADS + hh] + d[di * HEADS + hh];
        v = v > 0.f ? v : 0.2f * v;                     // LeakyReLU(0.2)
        e[(size_t)i * HEADS + hh] = v;
        atomicMax(&mEnc[di * HEADS + hh], ordEnc(v));
    }
}

// ------------------------------- edge pass 2: numerator + denominator sums
__global__ void edge_softmax_num(const int* __restrict__ dstA,
                                 float* __restrict__ e, const unsigned* __restrict__ mEnc,
                                 float* __restrict__ den, int E, int N) {
    int i = blockIdx.x * blockDim.x + threadIdx.x;
    if (i >= E + N) return;
    int di = (i < E) ? dstA[i] : (i - E);
#pragma unroll
    for (int hh = 0; hh < HEADS; ++hh) {
        float m = ordDec(mEnc[di * HEADS + hh]);
        float w = __expf(e[(size_t)i * HEADS + hh] - m);
        e[(size_t)i * HEADS + hh] = w;
        atomicAddF32(&den[di * HEADS + hh], w);
    }
}

// --------------------------------------- edge pass 3: weighted scatter-add
__global__ void edge_scatter(const int* __restrict__ srcA, const int* __restrict__ dstA,
                             const float* __restrict__ w, const float* __restrict__ den,
                             const float* __restrict__ hf, float* __restrict__ out,
                             int E, int N) {
    int i = blockIdx.x;                 // edge (or self loop) id
    int t = threadIdx.x;                // channel 0..255, head = t>>6
    int si = (i < E) ? srcA[i] : (i - E);
    int di = (i < E) ? dstA[i] : (i - E);
    int hh = t >> 6;
    float alpha = w[(size_t)i * HEADS + hh] / den[di * HEADS + hh];
    atomicAddF32(&out[(size_t)di * HC + t], alpha * hf[(size_t)si * HC + t]);
}

// --------------------------------------------------------------- bias add
__global__ void bias_add(float* __restrict__ x, const float* __restrict__ b, int total, int C) {
    int i = blockIdx.x * blockDim.x + threadIdx.x;
    if (i < total) x[i] += b[i & (C - 1)];
}

// ------------------------------------- layer-2: mean over heads + bias
__global__ void head_mean_bias(const float* __restrict__ acc, const float* __restrict__ b,
                               float* __restrict__ out, int N) {
    int i = blockIdx.x * blockDim.x + threadIdx.x;
    if (i >= N * CH) return;
    int n = i >> 6, c = i & 63;
    const float* r = acc + (size_t)n * HC;
    out[i] = 0.25f * (r[c] + r[CH + c] + r[2 * CH + c] + r[3 * CH + c]) + b[c];
}

// ------------------------------------------------ GraphNorm reductions
__global__ void gn_colsum(const float* __restrict__ x, float* __restrict__ colsum,
                          int N, int C) {
    int t = threadIdx.x;
    int ch = t & (C - 1);
    int rsub = t / C;
    int rpi = 256 / C;                     // rows per 256-thread iteration
    float acc = 0.f;
    for (int row = blockIdx.x * rpi + rsub; row < N; row += gridDim.x * rpi)
        acc += x[(size_t)row * C + ch];
    atomicAddF32(&colsum[ch], acc);
}
__global__ void gn_varsum(const float* __restrict__ x, const float* __restrict__ colsum,
                          const float* __restrict__ ga, float* __restrict__ varsum,
                          int N, int C) {
    int t = threadIdx.x;
    int ch = t & (C - 1);
    int rsub = t / C;
    int rpi = 256 / C;
    float mu = colsum[ch] * (1.f / (float)N);
    float a = ga[ch];
    float acc = 0.f;
    for (int row = blockIdx.x * rpi + rsub; row < N; row += gridDim.x * rpi) {
        float y = x[(size_t)row * C + ch] - a * mu;
        acc += y * y;
    }
    atomicAddF32(&varsum[ch], acc);
}
__global__ void gn_apply(const float* __restrict__ x,
                         const float* __restrict__ colsum, const float* __restrict__ varsum,
                         const float* __restrict__ ga, const float* __restrict__ gw,
                         const float* __restrict__ gb,
                         float* __restrict__ outf, __bf16* __restrict__ outb,
                         int N, int C, float slope) {
    int i = blockIdx.x * blockDim.x + threadIdx.x;
    if (i >= N * C) return;
    int ch = i & (C - 1);
    float invN = 1.f / (float)N;
    float mu  = colsum[ch] * invN;
    float var = varsum[ch] * invN;
    float y = x[i] - ga[ch] * mu;
    float v = gw[ch] * y * rsqrtf(var + 1e-5f) + gb[ch];
    v = v > 0.f ? v : slope * v;          // LeakyReLU(0.01)
    outf[i] = v;
    if (outb) outb[i] = (__bf16)v;
}

// ------------------------------------------------------------------- MLP
__global__ void mlp_head(const float* __restrict__ x,
                         const float* __restrict__ mW0, const float* __restrict__ mb0,
                         const float* __restrict__ mW1, const float* __restrict__ mb1,
                         const float* __restrict__ mW2, const float* __restrict__ mb2,
                         float* __restrict__ out, int N) {
    __shared__ float w0[64 * 32], w1[32 * 16], w2[16 * 2], b0s[32], b1s[16], b2s[2];
    int t = threadIdx.x;
    for (int i = t; i < 64 * 32; i += 256) w0[i] = mW0[i];
    for (int i = t; i < 32 * 16; i += 256) w1[i] = mW1[i];
    if (t < 32) { b0s[t] = mb0[t]; w2[t] = mW2[t]; }
    if (t < 16) b1s[t] = mb1[t];
    if (t < 2)  b2s[t] = mb2[t];
    __syncthreads();
    int n = blockIdx.x * 256 + t;
    if (n >= N) return;
    float in[64];
#pragma unroll
    for (int k = 0; k < 64; ++k) in[k] = x[(size_t)n * 64 + k];
    float v1[32];
#pragma unroll
    for (int j = 0; j < 32; ++j) {
        float a = b0s[j];
        for (int k = 0; k < 64; ++k) a += in[k] * w0[k * 32 + j];
        v1[j] = fmaxf(a, 0.f);
    }
    float v2[16];
#pragma unroll
    for (int j = 0; j < 16; ++j) {
        float a = b1s[j];
        for (int k = 0; k < 32; ++k) a += v1[k] * w1[k * 16 + j];
        v2[j] = fmaxf(a, 0.f);
    }
    float o0 = b2s[0], o1 = b2s[1];
#pragma unroll
    for (int k = 0; k < 16; ++k) { o0 += v2[k] * w2[k * 2]; o1 += v2[k] * w2[k * 2 + 1]; }
    out[(size_t)n * 2 + 0] = o0;
    out[(size_t)n * 2 + 1] = o1;
}

// ===========================================================================
extern "C" void kernel_launch(void* const* d_in, const int* in_sizes, int n_in,
                              void* d_out, int out_size, void* d_ws, size_t ws_size,
                              hipStream_t stream) {
    const float* x  = (const float*)d_in[0];
    const int*   ei = (const int*)d_in[1];
    const int N = in_sizes[0] / HC;         // 50000
    const int E = in_sizes[1] / 2;          // 800000
    const int* srcA = ei;
    const int* dstA = ei + E;
    const int EN = E + N;                   // edges incl. self loops

    // ---- workspace layout (256B aligned) ----
    char* wsp = (char*)d_ws;
    size_t off = 0;
    auto alloc = [&](size_t bytes) -> void* {
        void* p = wsp + off;
        off += (bytes + 255) & ~(size_t)255;
        return p;
    };
    __bf16*   actB  = (__bf16*)  alloc((size_t)N * HC * 2);  // bf16 activations (GEMM A)
    __bf16*   WtB   = (__bf16*)  alloc((size_t)HC * HC * 2); // transposed bf16 weights
    float*    hlin  = (float*)   alloc((size_t)N * HC * 4);  // GEMM out / next-layer f32
    float*    hacc  = (float*)   alloc((size_t)N * HC * 4);  // scatter accumulator
    float*    sbuf  = (float*)   alloc((size_t)N * HEADS * 4);
    float*    dbuf  = (float*)   alloc((size_t)N * HEADS * 4);
    unsigned* mEnc  = (unsigned*)alloc((size_t)N * HEADS * 4);
    float*    den   = (float*)   alloc((size_t)N * HEADS * 4);
    float*    ebuf  = (float*)   alloc((size_t)EN * HEADS * 4);
    float*    colsum= (float*)   alloc(256 * 4);
    float*    varsum= (float*)   alloc(256 * 4);
    float*    tmp64 = ebuf;                 // reuse: [N,64] needed only after edge phase

    const dim3 gemmGrid(N / 16, HC / 64);   // 3125 x 4, 128 threads (4 waves)
    const int TB = 256;
    const int nhcBlk = (N * HC + TB - 1) / TB;
    const int enBlk  = (EN + TB - 1) / TB;

    // x -> bf16 once
    cvt_f32_bf16<<<nhcBlk, TB, 0, stream>>>(x, actB, N * HC);

    for (int layer = 0; layer < 3; ++layer) {
        const int base = 2 + layer * 7;
        const float* W   = (const float*)d_in[base + 0];
        const float* as_ = (const float*)d_in[base + 1];
        const float* ad_ = (const float*)d_in[base + 2];
        const float* b   = (const float*)d_in[base + 3];
        const float* gw  = (const float*)d_in[base + 4];
        const float* gb  = (const float*)d_in[base + 5];
        const float* ga  = (const float*)d_in[base + 6];

        // 1) weights -> transposed bf16; GEMM h = act @ W  (WMMA)
        wt_bf16<<<(HC * HC + TB - 1) / TB, TB, 0, stream>>>(W, WtB, HC, HC);
        gemm_bf16_wmma<<<gemmGrid, 128, 0, stream>>>(actB, WtB, hlin, N);

        // 2) per-node attention halves
        node_attn_sd<<<N, HC, 0, stream>>>(hlin, as_, ad_, sbuf, dbuf);

        // 3) init segment state + accumulator
        fill_u32<<<(N * HEADS + TB - 1) / TB, TB, 0, stream>>>(mEnc, ORD_NEG_INF, N * HEADS);
        fill_f32<<<(N * HEADS + TB - 1) / TB, TB, 0, stream>>>(den, 0.f, N * HEADS);
        fill_f32<<<nhcBlk, TB, 0, stream>>>(hacc, 0.f, N * HC);

        // 4) segment softmax + scatter
        edge_logits_max<<<enBlk, TB, 0, stream>>>(srcA, dstA, sbuf, dbuf, ebuf, mEnc, E, N);
        edge_softmax_num<<<enBlk, TB, 0, stream>>>(dstA, ebuf, mEnc, den, E, N);
        edge_scatter<<<EN, HC, 0, stream>>>(srcA, dstA, ebuf, den, hlin, hacc, E, N);

        if (layer < 2) {
            // 5) + bias, GraphNorm(C=256), LeakyReLU(0.01) -> hlin (f32) + actB (bf16)
            bias_add<<<nhcBlk, TB, 0, stream>>>(hacc, b, N * HC, HC);
            fill_f32<<<1, 256, 0, stream>>>(colsum, 0.f, HC);
            fill_f32<<<1, 256, 0, stream>>>(varsum, 0.f, HC);
            gn_colsum<<<512, TB, 0, stream>>>(hacc, colsum, N, HC);
            gn_varsum<<<512, TB, 0, stream>>>(hacc, colsum, ga, varsum, N, HC);
            gn_apply<<<nhcBlk, TB, 0, stream>>>(hacc, colsum, varsum, ga, gw, gb,
                                                hlin, actB, N, HC, 0.01f);
        } else {
            // 5') mean over heads + bias, GraphNorm(C=64) in-place on tmp64
            head_mean_bias<<<(N * CH + TB - 1) / TB, TB, 0, stream>>>(hacc, b, tmp64, N);
            fill_f32<<<1, 256, 0, stream>>>(colsum, 0.f, CH);
            fill_f32<<<1, 256, 0, stream>>>(varsum, 0.f, CH);
            gn_colsum<<<512, TB, 0, stream>>>(tmp64, colsum, N, CH);
            gn_varsum<<<512, TB, 0, stream>>>(tmp64, colsum, ga, varsum, N, CH);
            gn_apply<<<(N * CH + TB - 1) / TB, TB, 0, stream>>>(tmp64, colsum, varsum,
                                                               ga, gw, gb, tmp64,
                                                               (__bf16*)nullptr, N, CH, 0.01f);
        }
    }

    // 6) MLP head -> d_out [N,2]
    mlp_head<<<(N + 255) / 256, 256, 0, stream>>>(
        tmp64,
        (const float*)d_in[23], (const float*)d_in[24],
        (const float*)d_in[25], (const float*)d_in[26],
        (const float*)d_in[27], (const float*)d_in[28],
        (float*)d_out, N);
}
